// Multiproto_60928406061320
// MI455X (gfx1250) — matrix operand order repaired and checked by reading, not compiled, not run on hardware
//
#include <hip/hip_runtime.h>

typedef _Float16 h16;
typedef unsigned int u32;
typedef __attribute__((ext_vector_type(16))) _Float16 v16h;
typedef __attribute__((ext_vector_type(8)))  _Float16 v8h;
typedef __attribute__((ext_vector_type(8)))  float    v8f;
typedef __attribute__((ext_vector_type(4)))  unsigned int v4u;
typedef __attribute__((ext_vector_type(8)))  int v8i;
typedef __attribute__((ext_vector_type(4)))  int v4i;

#define D_DIM 256
#define P_DIM 2000
#define P_PAD 2016   // padded proto count (multiple of 32)
#define NCLSS 100
#define CPAD 112     // padded class count (multiple of 16, 32B-aligned rows)
#define N_ROWS 51200
#define WAVES 2
#define CSTG 256     // per-wave stage width (holds h, later x_tilde)

static __device__ __forceinline__ v8f wmma16(v16h a, v16h b, v8f c) {
  return __builtin_amdgcn_wmma_f32_16x16x32_f16(false, a, false, b, (short)0, c, false, false);
}

// B fragment (32x16, f16) from global k-major storage: lane L holds N = nb + L%16, elements = K (table order)
static __device__ __forceinline__ v16h ldB(const h16* __restrict__ M, int ld, int kb, int nb, int lane) {
  int n = nb + (lane & 15), hh = (lane >> 4) * 8;
  v16h b;
#pragma unroll
  for (int i = 0; i < 8; ++i) { b[i] = M[(size_t)(kb + hh + i) * ld + n]; b[i + 8] = M[(size_t)(kb + 16 + hh + i) * ld + n]; }
  return b;
}

// B fragment from an LDS-staged [rows][width] k-major tile
static __device__ __forceinline__ v16h ldB_lds(const h16* S, int width, int kb, int nb, int lane) {
  int n = nb + (lane & 15), hh = (lane >> 4) * 8;
  v16h b;
#pragma unroll
  for (int i = 0; i < 8; ++i) { b[i] = S[(size_t)(kb + hh + i) * width + n]; b[i + 8] = S[(size_t)(kb + 16 + hh + i) * width + n]; }
  return b;
}

// A fragment (16x32, f16) from LDS row-major tile, row stride `srow` halves
static __device__ __forceinline__ v16h ldA_lds(const h16* S, int srow, int kb, int lane) {
  int r = lane & 15, hh = (lane >> 4) * 8;
  const v8h* p0 = (const v8h*)(S + r * srow + kb + hh);
  const v8h* p1 = (const v8h*)(S + r * srow + kb + 16 + hh);
  v8h lo = *p0, hi = *p1;
  v16h a;
#pragma unroll
  for (int i = 0; i < 8; ++i) { a[i] = lo[i]; a[i + 8] = hi[i]; }
  return a;
}

// A fragment from global f32 (samples), converting to f16
static __device__ __forceinline__ v16h ldA_gf32(const float* __restrict__ src, int ld, int row0, int kb, int lane) {
  int r = row0 + (lane & 15), hh = (lane >> 4) * 8;
  const float* p0 = src + (size_t)r * ld + kb + hh;
  const float* p1 = p0 + 16;
  v16h a;
#pragma unroll
  for (int i = 0; i < 8; ++i) { a[i] = (h16)p0[i]; a[i + 8] = (h16)p1[i]; }
  return a;
}

static __device__ __forceinline__ float rmax16(float x) {
#pragma unroll
  for (int m = 1; m < 16; m <<= 1) x = fmaxf(x, __shfl_xor(x, m, 32));
  return x;
}
static __device__ __forceinline__ float rsum16(float x) {
#pragma unroll
  for (int m = 1; m < 16; m <<= 1) x += __shfl_xor(x, m, 32);
  return x;
}

// ---- Tensor Data Mover: async 2D-tile load (tile_w x rows, strided rows) into compact LDS ----
// D# packing per CDNA5 ISA.
static __device__ __forceinline__ void tdm_load_tile(const h16* gaddr, u32 lds_off,
                                                     u32 tile_w, u32 rows, u32 row_stride) {
  unsigned long long ga = (unsigned long long)(size_t)gaddr;
  v4u g0;
  g0[0] = 1u;                                              // count=1, user descriptor
  g0[1] = lds_off;                                         // LDS byte address
  g0[2] = (u32)(ga & 0xFFFFFFFFull);                       // global_addr[31:0]
  g0[3] = (u32)((ga >> 32) & 0x1FFFFFFull) | (2u << 30);   // global_addr[56:32] | type=2
  v8i g1;
  g1[0] = (int)(1u << 16);                                 // data_size=2B; mask/flags=0
  g1[1] = (int)((row_stride & 0xFFFFu) << 16);             // tensor_dim0[15:0]
  g1[2] = (int)((row_stride >> 16) | (rows << 16));        // tensor_dim0[31:16] | tensor_dim1[15:0]
  g1[3] = (int)((rows >> 16) | (tile_w << 16));            // tensor_dim1[31:16] | tile_dim0
  g1[4] = (int)rows;                                       // tile_dim1 (tile_dim2=0)
  g1[5] = (int)row_stride;                                 // tensor_dim0_stride[31:0]
  g1[6] = 0;
  g1[7] = 0;
  v4i gz = {0, 0, 0, 0};
#if defined(__clang_major__) && (__clang_major__ >= 23)
  v8i gz8 = {0, 0, 0, 0, 0, 0, 0, 0};
  __builtin_amdgcn_tensor_load_to_lds(g0, g1, gz, gz, gz8, 0);
#else
  __builtin_amdgcn_tensor_load_to_lds(g0, g1, gz, gz, 0);
#endif
}

// ---------------- prep: f32 -> f16 weight copies / transposes / pads, ||p||^2 ----------------
__global__ void prep_kernel(const float* __restrict__ hidden_w, const float* __restrict__ prototypes,
                            const float* __restrict__ output_w,
                            h16* __restrict__ WT, h16* __restrict__ protoF, h16* __restrict__ protoT,
                            h16* __restrict__ outWT, float* __restrict__ p2) {
  int tid = blockIdx.x * blockDim.x + threadIdx.x;
  int nth = gridDim.x * blockDim.x;
  for (int i = tid; i < D_DIM * D_DIM; i += nth) {              // WT[k][o] = hidden_w[o][k]
    int k = i / D_DIM, o = i % D_DIM;
    WT[i] = (h16)hidden_w[(size_t)o * D_DIM + k];
  }
  for (int i = tid; i < P_PAD * D_DIM; i += nth) {              // protoF: row-major, padded rows
    int p = i / D_DIM, d = i % D_DIM;
    protoF[i] = (p < P_DIM) ? (h16)prototypes[(size_t)p * D_DIM + d] : (h16)0.f;
  }
  for (int i = tid; i < D_DIM * P_PAD; i += nth) {              // protoT[k][p]: k-major, padded cols
    int k = i / P_PAD, p = i % P_PAD;
    protoT[i] = (p < P_DIM) ? (h16)prototypes[(size_t)p * D_DIM + k] : (h16)0.f;
  }
  for (int i = tid; i < 2 * D_DIM * CPAD; i += nth) {           // outWT[k][c], classes padded to 112
    int k = i / CPAD, c = i % CPAD;
    outWT[i] = (c < NCLSS) ? (h16)output_w[(size_t)c * (2 * D_DIM) + k] : (h16)0.f;
  }
  for (int p = tid; p < P_DIM; p += nth) {                      // squared norms in f32
    float s = 0.f;
    const float* row = prototypes + (size_t)p * D_DIM;
    for (int d = 0; d < D_DIM; ++d) { float x = row[d]; s += x * x; }
    p2[p] = s;
  }
}

// ---------------- fused: hidden+ReLU -> flash-softmax over prototypes -> x_tilde -> output ----------------
__global__ __launch_bounds__(WAVES * 32) void fused_kernel(
    const float* __restrict__ samples, const float* __restrict__ hidden_b,
    const h16* __restrict__ WT, const h16* __restrict__ protoF, const h16* __restrict__ protoT,
    const h16* __restrict__ outWT, const float* __restrict__ p2, float* __restrict__ out) {
  // per-wave [16][256] stage (h, then x_tilde) + [16][32] q stage; 2x 16KB double-buffered TDM tiles
  __shared__ h16 lds[WAVES * (16 * CSTG + 16 * 32) + 2 * 8192];
  const int tid = threadIdx.x, lane = tid & 31, wv = tid >> 5;
  h16* cstage = lds + wv * (16 * CSTG + 16 * 32);
  h16* qstage = cstage + 16 * CSTG;
  h16* buf[2];
  buf[0] = lds + WAVES * (16 * CSTG + 16 * 32);
  buf[1] = buf[0] + 8192;
  const u32 boff0 = (u32)(size_t)(void*)buf[0];   // LDS aperture: low 32 bits = LDS byte offset
  const u32 boff1 = (u32)(size_t)(void*)buf[1];
  const int R0 = (blockIdx.x * WAVES + wv) * 16;
  const int l15 = lane & 15, hf = lane >> 4;
  v8f zero = {};

  // ---- Phase 1: h = relu(x @ W^T + b) via TDM-staged WT slabs [256k x 32] ----
  if (wv == 0) tdm_load_tile(WT, boff0, 32u, 256u, D_DIM);   // slab j=0 (overlaps sample loads)
  v16h xfrag[8];
#pragma unroll
  for (int k = 0; k < 8; ++k) xfrag[k] = ldA_gf32(samples, D_DIM, R0, 32 * k, lane);
#pragma unroll
  for (int j = 0; j < 8; ++j) {
    if (wv == 0) __builtin_amdgcn_s_wait_tensorcnt(0);
    __syncthreads();                               // publish buf[j&1]; other buffer free
    if (wv == 0 && j < 7) tdm_load_tile(WT + 32 * (j + 1), (j & 1) ? boff0 : boff1, 32u, 256u, D_DIM);
    const h16* wt = buf[j & 1];
    v8f c0 = zero, c1 = zero;
#pragma unroll
    for (int k = 0; k < 8; ++k) {
      c0 = wmma16(xfrag[k], ldB_lds(wt, 32, 32 * k, 0, lane), c0);
      c1 = wmma16(xfrag[k], ldB_lds(wt, 32, 32 * k, 16, lane), c1);
    }
    float b0 = hidden_b[32 * j + l15], b1 = hidden_b[32 * j + 16 + l15];
#pragma unroll
    for (int v = 0; v < 8; ++v) {
      int row = v + 8 * hf;
      cstage[row * CSTG + 32 * j + l15]      = (h16)fmaxf(c0[v] + b0, 0.f);
      cstage[row * CSTG + 32 * j + 16 + l15] = (h16)fmaxf(c1[v] + b1, 0.f);
    }
  }

  // ---- Phase 2: online softmax over P (s = 2*h.p - |p|^2), accumulate x_tilde ----
  float m[8], l[8];
#pragma unroll
  for (int v = 0; v < 8; ++v) { m[v] = -1e30f; l[v] = 0.f; }
  v8f acc[16];
#pragma unroll
  for (int t = 0; t < 16; ++t) acc[t] = zero;

  if (wv == 0) tdm_load_tile(protoT, boff0, 32u, 256u, P_PAD);  // chunk 0
  for (int pb = 0, c = 0; pb < P_DIM; pb += 32, ++c) {
    if (wv == 0) __builtin_amdgcn_s_wait_tensorcnt(0);
    __syncthreads();
    if (wv == 0 && pb + 32 < P_DIM)
      tdm_load_tile(protoT + pb + 32, (c & 1) ? boff0 : boff1, 32u, 256u, P_PAD);
    const h16* pt = buf[c & 1];

    __builtin_prefetch(protoF + (size_t)(pb + 32 + lane) * D_DIM, 0, 1);

    v8f g0 = zero, g1 = zero;
#pragma unroll
    for (int k = 0; k < 8; ++k) {
      v16h a = ldA_lds(cstage, CSTG, 32 * k, lane);
      g0 = wmma16(a, ldB_lds(pt, 32, 32 * k, 0, lane), g0);
      g1 = wmma16(a, ldB_lds(pt, 32, 32 * k, 16, lane), g1);
    }
    float p2a = p2[pb + l15];
    float p2b = (pb + 16 < P_DIM) ? p2[pb + 16 + l15] : 1e30f;  // padded scores -> -inf

    float e0[8], e1[8], scale[8];
#pragma unroll
    for (int v = 0; v < 8; ++v) {
      float s0 = 2.f * g0[v] - p2a;
      float s1 = 2.f * g1[v] - p2b;
      float tm = rmax16(fmaxf(s0, s1));
      float mn = fmaxf(m[v], tm);
      float sc = __expf(m[v] - mn);
      e0[v] = __expf(s0 - mn);
      e1[v] = __expf(s1 - mn);
      l[v] = l[v] * sc + rsum16(e0[v] + e1[v]);
      m[v] = mn;
      scale[v] = sc;
    }
#pragma unroll
    for (int t = 0; t < 16; ++t)
#pragma unroll
      for (int v = 0; v < 8; ++v) acc[t][v] *= scale[v];

    // q tile: C-layout -> LDS -> A-layout (f16, values in [0,1])
#pragma unroll
    for (int v = 0; v < 8; ++v) {
      int row = v + 8 * hf;
      qstage[row * 32 + l15]      = (h16)e0[v];
      qstage[row * 32 + 16 + l15] = (h16)e1[v];
    }
    v16h aq = ldA_lds(qstage, 32, 0, lane);
#pragma unroll
    for (int t = 0; t < 16; ++t)
      acc[t] = wmma16(aq, ldB(protoF, D_DIM, pb, 16 * t, lane), acc[t]);
  }

  // ---- Phase 3a: co[ct] = h @ outW^T[0:256] via TDM-staged outWT slabs [512k x 16] ----
  __syncthreads();                                  // last proto buffer readers done
  if (wv == 0) tdm_load_tile(outWT, boff0, 16u, 512u, CPAD);
  v8f co[7];
#pragma unroll
  for (int ct = 0; ct < 7; ++ct) {
    if (wv == 0) __builtin_amdgcn_s_wait_tensorcnt(0);
    __syncthreads();
    if (wv == 0 && ct < 6) tdm_load_tile(outWT + 16 * (ct + 1), (ct & 1) ? boff0 : boff1, 16u, 512u, CPAD);
    const h16* ot = buf[ct & 1];
    v8f c = zero;
#pragma unroll
    for (int k = 0; k < 8; ++k)
      c = wmma16(ldA_lds(cstage, CSTG, 32 * k, lane), ldB_lds(ot, 16, 32 * k, 0, lane), c);
    co[ct] = c;
  }

  // normalize x_tilde and overwrite the (wave-private) stage with it
#pragma unroll
  for (int v = 0; v < 8; ++v) l[v] = 1.f / l[v];
#pragma unroll
  for (int t = 0; t < 16; ++t)
#pragma unroll
    for (int v = 0; v < 8; ++v) {
      int row = v + 8 * hf;
      cstage[row * CSTG + 16 * t + l15] = (h16)(acc[t][v] * l[v]);
    }

  // ---- Phase 3b: co[ct] += x_tilde @ outW^T[256:512]; store ----
  __syncthreads();
  if (wv == 0) tdm_load_tile(outWT, boff0, 16u, 512u, CPAD);
#pragma unroll
  for (int ct = 0; ct < 7; ++ct) {
    if (wv == 0) __builtin_amdgcn_s_wait_tensorcnt(0);
    __syncthreads();
    if (wv == 0 && ct < 6) tdm_load_tile(outWT + 16 * (ct + 1), (ct & 1) ? boff0 : boff1, 16u, 512u, CPAD);
    const h16* ot = buf[ct & 1];
    v8f c = co[ct];
#pragma unroll
    for (int k = 0; k < 8; ++k)
      c = wmma16(ldA_lds(cstage, CSTG, 32 * k, lane), ldB_lds(ot, 16, 256 + 32 * k, 0, lane), c);
    int col = 16 * ct + l15;
    if (col < NCLSS) {
#pragma unroll
      for (int v = 0; v < 8; ++v) {
        int row = R0 + v + 8 * hf;
        out[(size_t)row * NCLSS + col] = c[v];
      }
    }
  }
}

extern "C" void kernel_launch(void* const* d_in, const int* in_sizes, int n_in,
                              void* d_out, int out_size, void* d_ws, size_t ws_size,
                              hipStream_t stream) {
  const float* samples    = (const float*)d_in[0];
  const float* hidden_w   = (const float*)d_in[1];
  const float* hidden_b   = (const float*)d_in[2];
  const float* prototypes = (const float*)d_in[3];
  const float* output_w   = (const float*)d_in[4];
  char* ws = (char*)d_ws;
  h16*   WT     = (h16*)(ws + 0);          // 256*256 f16        = 131,072 B
  h16*   protoF = (h16*)(ws + 131072);     // 2016*256 f16       = 1,032,192 B
  h16*   protoT = (h16*)(ws + 1163264);    // 256*2016 f16       = 1,032,192 B
  h16*   outWT  = (h16*)(ws + 2195456);    // 512*112 f16        = 114,688 B
  float* p2     = (float*)(ws + 2310144);  // 2000 f32           = 8,000 B

  prep_kernel<<<512, 256, 0, stream>>>(hidden_w, prototypes, output_w, WT, protoF, protoT, outWT, p2);
  fused_kernel<<<N_ROWS / (WAVES * 16), WAVES * 32, 0, stream>>>(
      samples, hidden_b, WT, protoF, protoT, outWT, p2, (float*)d_out);
}
